// MatrixGPT_9483287789980
// MI455X (gfx1250) — matrix operand order repaired and checked by reading, not compiled, not run on hardware
//
#include <hip/hip_runtime.h>

// ---------------------------------------------------------------------------
// MatrixGPT forward on MI455X (gfx1250): wave32 + v_wmma_f32_16x16x32_f16
// ---------------------------------------------------------------------------
// Shapes: B=32 T=1024 D=512 k=128 L=6 V=2048, BT = 32768 tokens.

#define BT_TOK   32768
#define DIM      512
#define KMAT     128
#define NLAYER   6
#define VOCAB    2048
#define EPS_RMS  1.1920929e-07f
#define SOFTCAP  30.0f

typedef _Float16 v16h __attribute__((ext_vector_type(16)));
typedef _Float16 v8h  __attribute__((ext_vector_type(8)));
typedef float    v8f  __attribute__((ext_vector_type(8)));

union AFrag { v16h v; v8h h[2]; };

// A fragment: 16 rows (tokens) x 32 K.  lane&15 = row; lane>=16 shifts K by 8.
// Per ISA 7.12.2 (16-bit A 16x32): lane<16 holds K {0..7,16..23}, lane>=16 {8..15,24..31}.
__device__ __forceinline__ v16h load_frag_a(const _Float16* __restrict__ base, int ld,
                                            int kbase, int lane) {
  const _Float16* p = base + (lane & 15) * ld + kbase + ((lane >> 4) << 3);
  AFrag f;
  f.h[0] = *(const v8h*)p;
  f.h[1] = *(const v8h*)(p + 16);
  return f.v;
}

// B fragment: 32 K x 16 cols.  lane&15 = col; lanes 0-15 K {0..7,16..23}, lanes 16-31 K {8..15,24..31}
// (same K order as A).  Source is row-major (col, K) i.e. the weight is used transposed,
// which matches einsum('btd,fd->btf') / ('bth,dh->btd') / ('btd,vd->btv') directly.
__device__ __forceinline__ v16h load_frag_b(const _Float16* __restrict__ base, int ld,
                                            int kbase, int lane) {
  const _Float16* p = base + (lane & 15) * ld + kbase + ((lane >> 4) << 3);
  AFrag f;
  f.h[0] = *(const v8h*)p;
  f.h[1] = *(const v8h*)(p + 16);
  return f.v;
}

// ---------------------------------------------------------------------------
// 1) embedding gather + rmsnorm -> x (fp32)
// ---------------------------------------------------------------------------
__global__ __launch_bounds__(256) void embed_rms_kernel(const int* __restrict__ ids,
                                                        const float* __restrict__ emb,
                                                        float* __restrict__ x) {
  int token = blockIdx.x;
  const float* row = emb + (size_t)ids[token] * DIM;
  int i = threadIdx.x * 2;
  float2 v = *(const float2*)(row + i);
  float ss = v.x * v.x + v.y * v.y;
  for (int m = 1; m < 32; m <<= 1) ss += __shfl_xor(ss, m, 32);
  __shared__ float red[8];
  int lane = threadIdx.x & 31, wv = threadIdx.x >> 5;
  if (lane == 0) red[wv] = ss;
  __syncthreads();
  float tot = red[0];
#pragma unroll
  for (int j = 1; j < 8; ++j) tot += red[j];
  float s = rsqrtf(tot * (1.0f / DIM) + EPS_RMS);
  float* out = x + (size_t)token * DIM;
  out[i] = v.x * s;
  out[i + 1] = v.y * s;
}

// ---------------------------------------------------------------------------
// 2) y = rmsnorm(x) -> f16
// ---------------------------------------------------------------------------
__global__ __launch_bounds__(256) void rms_to_half_kernel(const float* __restrict__ x,
                                                          _Float16* __restrict__ y) {
  int token = blockIdx.x;
  const float* row = x + (size_t)token * DIM;
  int i = threadIdx.x * 2;
  float2 v = *(const float2*)(row + i);
  float ss = v.x * v.x + v.y * v.y;
  for (int m = 1; m < 32; m <<= 1) ss += __shfl_xor(ss, m, 32);
  __shared__ float red[8];
  int lane = threadIdx.x & 31, wv = threadIdx.x >> 5;
  if (lane == 0) red[wv] = ss;
  __syncthreads();
  float tot = red[0];
#pragma unroll
  for (int j = 1; j < 8; ++j) tot += red[j];
  float s = rsqrtf(tot * (1.0f / DIM) + EPS_RMS);
  union { _Float16 q[2]; unsigned u; } pk;
  pk.q[0] = (_Float16)(v.x * s);
  pk.q[1] = (_Float16)(v.y * s);
  *(unsigned*)(y + (size_t)token * DIM + i) = pk.u;
}

// ---------------------------------------------------------------------------
// 3) fp32 -> f16 weight/embedding conversion
// ---------------------------------------------------------------------------
__global__ void f32_to_f16_kernel(const float* __restrict__ in, _Float16* __restrict__ out, int n) {
  int i = blockIdx.x * blockDim.x + threadIdx.x;
  if (i < n) out[i] = (_Float16)in[i];
}

// ---------------------------------------------------------------------------
// 4) WMMA GEMM  C[M,N] = A[M,K] * Bw[N,K]^T.
//    Block = 256 thr = 8 waves; block tile 64 rows x 128 cols; wave tile 16x64.
//    Software-pipelined and unrolled x2 with ping-pong fragment sets so the
//    rotation is free (no v_mov shuffling), and waits on loadcnt stay partial.
//    mode Cout!=null : write fp32 C (ld = ldn)
//    else           : x[row,col] += scale[col]*C  (fused residual of layer)
// ---------------------------------------------------------------------------
__global__ __launch_bounds__(256) void gemm_wmma_kernel(const _Float16* __restrict__ A,
                                                        const _Float16* __restrict__ Bw,
                                                        int Kdim, float* __restrict__ Cout,
                                                        float* __restrict__ Xres,
                                                        const float* __restrict__ scale,
                                                        int ldn) {
  int lane = threadIdx.x & 31;
  int w = threadIdx.x >> 5;
  int row0 = blockIdx.x * 64 + (w & 3) * 16;
  int col0 = blockIdx.y * 128 + (w >> 2) * 64;
  const _Float16* Abase = A + (size_t)row0 * Kdim;
  const _Float16* Bb[4];
#pragma unroll
  for (int j = 0; j < 4; ++j) Bb[j] = Bw + (size_t)(col0 + 16 * j) * Kdim;

  v8f acc[4];
#pragma unroll
  for (int j = 0; j < 4; ++j) acc[j] = (v8f){0.f, 0.f, 0.f, 0.f, 0.f, 0.f, 0.f, 0.f};

  v16h af0, af1, bf0[4], bf1[4];
  af0 = load_frag_a(Abase, Kdim, 0, lane);
#pragma unroll
  for (int j = 0; j < 4; ++j) bf0[j] = load_frag_b(Bb[j], Kdim, 0, lane);

  int kb = 0;
  for (; kb + 64 < Kdim; kb += 64) {
    af1 = load_frag_a(Abase, Kdim, kb + 32, lane);
#pragma unroll
    for (int j = 0; j < 4; ++j) bf1[j] = load_frag_b(Bb[j], Kdim, kb + 32, lane);
#pragma unroll
    for (int j = 0; j < 4; ++j)
      acc[j] = __builtin_amdgcn_wmma_f32_16x16x32_f16(false, af0, false, bf0[j],
                                                      (short)0, acc[j], false, false);
    af0 = load_frag_a(Abase, Kdim, kb + 64, lane);
#pragma unroll
    for (int j = 0; j < 4; ++j) bf0[j] = load_frag_b(Bb[j], Kdim, kb + 64, lane);
#pragma unroll
    for (int j = 0; j < 4; ++j)
      acc[j] = __builtin_amdgcn_wmma_f32_16x16x32_f16(false, af1, false, bf1[j],
                                                      (short)0, acc[j], false, false);
  }
  // tail: the last 64 K values (set0 already holds kb)
  af1 = load_frag_a(Abase, Kdim, kb + 32, lane);
#pragma unroll
  for (int j = 0; j < 4; ++j) bf1[j] = load_frag_b(Bb[j], Kdim, kb + 32, lane);
#pragma unroll
  for (int j = 0; j < 4; ++j)
    acc[j] = __builtin_amdgcn_wmma_f32_16x16x32_f16(false, af0, false, bf0[j],
                                                    (short)0, acc[j], false, false);
#pragma unroll
  for (int j = 0; j < 4; ++j)
    acc[j] = __builtin_amdgcn_wmma_f32_16x16x32_f16(false, af1, false, bf1[j],
                                                    (short)0, acc[j], false, false);

  int rbase = row0 + ((lane >> 4) << 3);
  int c = lane & 15;
  if (Cout) {
#pragma unroll
    for (int j = 0; j < 4; ++j) {
      int col = col0 + 16 * j + c;
#pragma unroll
      for (int r = 0; r < 8; ++r) Cout[(size_t)(rbase + r) * ldn + col] = acc[j][r];
    }
  } else {
#pragma unroll
    for (int j = 0; j < 4; ++j) {
      int col = col0 + 16 * j + c;
      float sc = scale[col];
#pragma unroll
      for (int r = 0; r < 8; ++r) {
        size_t off = (size_t)(rbase + r) * ldn + col;
        Xres[off] += sc * acc[j][r];
      }
    }
  }
}

// ---------------------------------------------------------------------------
// 5) fused: Frobenius-normalize 2x2, sequential prefix product over T,
//    h = S @ v, write h as f16.  One thread per (b, kk) chain.
//    The load stream is data-independent; prefetch 8 timesteps ahead
//    (global_prefetch_b8) so the 2x2-product chain never waits on HBM/L2.
// ---------------------------------------------------------------------------
__global__ __launch_bounds__(256) void scan_kernel(const float* __restrict__ M,
                                                   const float* __restrict__ vvec,
                                                   _Float16* __restrict__ h) {
  int tid = blockIdx.x * 256 + threadIdx.x;   // 0 .. 4095
  int b = tid >> 7;
  int kk = tid & 127;
  float v0 = vvec[kk * 2 + 0];
  float v1 = vvec[kk * 2 + 1];
  float p00 = 1.f, p01 = 0.f, p10 = 0.f, p11 = 1.f;  // identity
  const float* mp = M + (size_t)b * 1024 * DIM + kk * 4;
  _Float16* hp = h + (size_t)b * 1024 * (2 * KMAT) + kk * 2;
  for (int t = 0; t < 1024; ++t) {
    __builtin_prefetch(mp + 8 * DIM, 0, 3);
    float4 m = *(const float4*)mp;
    mp += DIM;
    float nrm = sqrtf(m.x * m.x + m.y * m.y + m.z * m.z + m.w * m.w);
    float inv = 1.0f / fmaxf(nrm, 1e-6f);
    float a00 = m.x * inv, a01 = m.y * inv, a10 = m.z * inv, a11 = m.w * inv;
    float n00 = a00 * p00 + a01 * p10;
    float n01 = a00 * p01 + a01 * p11;
    float n10 = a10 * p00 + a11 * p10;
    float n11 = a10 * p01 + a11 * p11;
    p00 = n00; p01 = n01; p10 = n10; p11 = n11;
    union { _Float16 q[2]; unsigned u; } pk;
    pk.q[0] = (_Float16)(n00 * v0 + n01 * v1);
    pk.q[1] = (_Float16)(n10 * v0 + n11 * v1);
    *(unsigned*)hp = pk.u;
    hp += 2 * KMAT;
  }
}

// ---------------------------------------------------------------------------
// 6) per-token softcapped target logit: zt = 30*tanh((y . emb[tgt])/30)
// ---------------------------------------------------------------------------
__global__ __launch_bounds__(256) void target_logit_kernel(const _Float16* __restrict__ y,
                                                           const _Float16* __restrict__ emb,
                                                           const int* __restrict__ tgt,
                                                           float* __restrict__ zt) {
  int wv = threadIdx.x >> 5, lane = threadIdx.x & 31;
  int token = blockIdx.x * 8 + wv;
  int t = tgt[token];
  const _Float16* yp = y + (size_t)token * DIM + lane * 16;
  const _Float16* ep = emb + (size_t)t * DIM + lane * 16;
  float d = 0.f;
#pragma unroll
  for (int i = 0; i < 16; ++i) d += (float)yp[i] * (float)ep[i];
  for (int m = 1; m < 32; m <<= 1) d += __shfl_xor(d, m, 32);
  if (lane == 0) zt[token] = SOFTCAP * tanhf(d * (1.0f / SOFTCAP));
}

// ---------------------------------------------------------------------------
// 7) loss: per 16-token block, WMMA logits over all V tiles with online
//    logsumexp (softcap applied per element); partial = sum_rows (LSE - z_tgt).
//    A tile staged in LDS (ds_load_b128); B fragments ping-pong (unroll x2).
// ---------------------------------------------------------------------------
__global__ __launch_bounds__(32) void loss_kernel(const _Float16* __restrict__ y,
                                                  const _Float16* __restrict__ emb,
                                                  const float* __restrict__ zt,
                                                  float* __restrict__ partials) {
  __shared__ __align__(16) _Float16 As[16 * DIM];
  int lane = threadIdx.x;
  int row0 = blockIdx.x * 16;
  const _Float16* src = y + (size_t)row0 * DIM;
  for (int i = lane * 8; i < 16 * DIM; i += 32 * 8) *(v8h*)(As + i) = *(const v8h*)(src + i);
  __syncthreads();

  float mrun[8], srun[8];
#pragma unroll
  for (int r = 0; r < 8; ++r) { mrun[r] = -1e30f; srun[r] = 0.0f; }

  for (int ct = 0; ct < VOCAB; ct += 16) {
    const _Float16* Bb = emb + (size_t)ct * DIM;
    v8f acc = (v8f){0.f, 0.f, 0.f, 0.f, 0.f, 0.f, 0.f, 0.f};
    v16h bf0 = load_frag_b(Bb, DIM, 0, lane);
    int kb = 0;
    for (; kb + 64 < DIM; kb += 64) {
      v16h bf1 = load_frag_b(Bb, DIM, kb + 32, lane);
      acc = __builtin_amdgcn_wmma_f32_16x16x32_f16(false, load_frag_a(As, DIM, kb, lane),
                                                   false, bf0, (short)0, acc, false, false);
      bf0 = load_frag_b(Bb, DIM, kb + 64, lane);
      acc = __builtin_amdgcn_wmma_f32_16x16x32_f16(false, load_frag_a(As, DIM, kb + 32, lane),
                                                   false, bf1, (short)0, acc, false, false);
    }
    {
      v16h bf1 = load_frag_b(Bb, DIM, kb + 32, lane);
      acc = __builtin_amdgcn_wmma_f32_16x16x32_f16(false, load_frag_a(As, DIM, kb, lane),
                                                   false, bf0, (short)0, acc, false, false);
      acc = __builtin_amdgcn_wmma_f32_16x16x32_f16(false, load_frag_a(As, DIM, kb + 32, lane),
                                                   false, bf1, (short)0, acc, false, false);
    }
#pragma unroll
    for (int r = 0; r < 8; ++r) {
      float z = SOFTCAP * tanhf(acc[r] * (1.0f / SOFTCAP));
      float mx = z;  // max over the 16 lanes holding this row (xor<16 stays in-half)
      mx = fmaxf(mx, __shfl_xor(mx, 1, 32));
      mx = fmaxf(mx, __shfl_xor(mx, 2, 32));
      mx = fmaxf(mx, __shfl_xor(mx, 4, 32));
      mx = fmaxf(mx, __shfl_xor(mx, 8, 32));
      float nm = fmaxf(mrun[r], mx);
      float e = __expf(z - nm);
      e += __shfl_xor(e, 1, 32);
      e += __shfl_xor(e, 2, 32);
      e += __shfl_xor(e, 4, 32);
      e += __shfl_xor(e, 8, 32);
      srun[r] = srun[r] * __expf(mrun[r] - nm) + e;
      mrun[r] = nm;
    }
  }

  float part = 0.f;
#pragma unroll
  for (int r = 0; r < 8; ++r) {
    float lse = mrun[r] + __logf(srun[r]);
    int row = row0 + r + ((lane >> 4) << 3);
    part += lse - zt[row];
  }
  part += __shfl_xor(part, 16, 32);  // lanes 0-15 rows 0-7, lanes 16-31 rows 8-15
  if (lane == 0) partials[blockIdx.x] = part;
}

// ---------------------------------------------------------------------------
// 8) deterministic final reduce: out = mean over BT tokens
// ---------------------------------------------------------------------------
__global__ __launch_bounds__(256) void reduce_kernel(const float* __restrict__ partials,
                                                     float* __restrict__ out) {
  __shared__ float s[256];
  float a = 0.f;
  for (int i = threadIdx.x; i < VOCAB; i += 256) a += partials[i];
  s[threadIdx.x] = a;
  __syncthreads();
  for (int st = 128; st > 0; st >>= 1) {
    if ((int)threadIdx.x < st) s[threadIdx.x] += s[threadIdx.x + st];
    __syncthreads();
  }
  if (threadIdx.x == 0) out[0] = s[0] / (float)BT_TOK;
}

// ---------------------------------------------------------------------------
// workspace layout (bytes)
// ---------------------------------------------------------------------------
static const size_t OFF_X    = 0;                                   // 64 MB fp32 residual
static const size_t OFF_Y    = OFF_X    + (size_t)BT_TOK * DIM * 4; // 32 MB f16 rmsnorm act
static const size_t OFF_M    = OFF_Y    + (size_t)BT_TOK * DIM * 2; // 64 MB fp32 M
static const size_t OFF_H    = OFF_M    + (size_t)BT_TOK * DIM * 4; // 16 MB f16 h
static const size_t OFF_WM   = OFF_H    + (size_t)BT_TOK * 2 * KMAT * 2; // 512 KB f16 Wm
static const size_t OFF_WP   = OFF_WM   + (size_t)DIM * DIM * 2;         // 256 KB f16 Wp
static const size_t OFF_EMB  = OFF_WP   + (size_t)DIM * 2 * KMAT * 2;    // 2 MB f16 emb
static const size_t OFF_ZT   = OFF_EMB  + (size_t)VOCAB * DIM * 2;       // 128 KB
static const size_t OFF_PART = OFF_ZT   + (size_t)BT_TOK * 4;            // 8 KB

extern "C" void kernel_launch(void* const* d_in, const int* in_sizes, int n_in,
                              void* d_out, int out_size, void* d_ws, size_t ws_size,
                              hipStream_t stream) {
  (void)in_sizes; (void)n_in; (void)out_size; (void)ws_size;
  const int*   ids = (const int*)d_in[0];
  const int*   tgt = (const int*)d_in[1];
  const float* emb = (const float*)d_in[2];   // (V, D)
  const float* wm  = (const float*)d_in[3];   // (L, 4k, D)
  const float* vv  = (const float*)d_in[4];   // (L, k, 2, 1)
  const float* wp  = (const float*)d_in[5];   // (L, D, 2k)
  const float* osc = (const float*)d_in[6];   // (L, D)

  char* ws = (char*)d_ws;
  float*     x    = (float*)(ws + OFF_X);
  _Float16*  y    = (_Float16*)(ws + OFF_Y);
  float*     Mbuf = (float*)(ws + OFF_M);
  _Float16*  hbuf = (_Float16*)(ws + OFF_H);
  _Float16*  wmh  = (_Float16*)(ws + OFF_WM);
  _Float16*  wph  = (_Float16*)(ws + OFF_WP);
  _Float16*  embh = (_Float16*)(ws + OFF_EMB);
  float*     ztb  = (float*)(ws + OFF_ZT);
  float*     part = (float*)(ws + OFF_PART);

  // embedding + first rmsnorm; emb->f16 once
  embed_rms_kernel<<<BT_TOK, 256, 0, stream>>>(ids, emb, x);
  f32_to_f16_kernel<<<(VOCAB * DIM + 255) / 256, 256, 0, stream>>>(emb, embh, VOCAB * DIM);

  for (int l = 0; l < NLAYER; ++l) {
    f32_to_f16_kernel<<<(DIM * DIM + 255) / 256, 256, 0, stream>>>(
        wm + (size_t)l * DIM * DIM, wmh, DIM * DIM);
    f32_to_f16_kernel<<<(DIM * 2 * KMAT + 255) / 256, 256, 0, stream>>>(
        wp + (size_t)l * DIM * 2 * KMAT, wph, DIM * 2 * KMAT);
    rms_to_half_kernel<<<BT_TOK, 256, 0, stream>>>(x, y);
    // M = y @ Wm^T : M=32768, N=512, K=512
    gemm_wmma_kernel<<<dim3(BT_TOK / 64, DIM / 128), 256, 0, stream>>>(
        y, wmh, DIM, Mbuf, nullptr, nullptr, DIM);
    // fused normalize + prefix scan + S@v -> h (f16)
    scan_kernel<<<16, 256, 0, stream>>>(Mbuf, vv + (size_t)l * 2 * KMAT, hbuf);
    // x += out_scale * (h @ Wp^T) : M=32768, N=512, K=256
    gemm_wmma_kernel<<<dim3(BT_TOK / 64, DIM / 128), 256, 0, stream>>>(
        hbuf, wph, 2 * KMAT, nullptr, x, osc + (size_t)l * DIM, DIM);
  }

  // final rmsnorm, softcapped logits + CE loss
  rms_to_half_kernel<<<BT_TOK, 256, 0, stream>>>(x, y);
  target_logit_kernel<<<BT_TOK / 8, 256, 0, stream>>>(y, embh, tgt, ztb);
  loss_kernel<<<BT_TOK / 16, 32, 0, stream>>>(y, embh, ztb, part);
  reduce_kernel<<<1, 256, 0, stream>>>(part, (float*)d_out);
}
